// HGT_21337397527227
// MI455X (gfx1250) — matrix operand order repaired and checked by reading, not compiled, not run on hardware
//
#include <hip/hip_runtime.h>

#define NMODEL   100000
#define NDATASET 20000
#define NTOT     120000
#define EDGES    500000
#define CC   128
#define HH   2
#define DD   64
#define LL   2
#define TT   2
#define RR   2

typedef __attribute__((ext_vector_type(16))) __bf16 v16bf;
typedef __attribute__((ext_vector_type(8)))  float  v8f;

__device__ __forceinline__ float gelu_exact(float x) {
    return 0.5f * x * (1.0f + erff(x * 0.7071067811865475f));
}
__device__ __forceinline__ unsigned fkey(float f) {
    unsigned u = __float_as_uint(f);
    return (u & 0x80000000u) ? ~u : (u | 0x80000000u);
}
__device__ __forceinline__ float funkey(unsigned k) {
    unsigned u = (k & 0x80000000u) ? (k & 0x7fffffffu) : ~k;
    return __uint_as_float(u);
}

// ---------------------------------------------------------------------------
// Weight prep: f32 -> bf16 elementwise
// ---------------------------------------------------------------------------
__global__ void cvt_bf16(const float* __restrict__ src, __bf16* __restrict__ dst, int n) {
    int i = blockIdx.x * blockDim.x + threadIdx.x;
    if (i < n) dst[i] = (__bf16)src[i];
}

// Fuse relation transform into projection weight:
//   Wout[k][h*64+e] = sum_d W[k][h*64+d] * rel[h][d][e]   (bf16 out)
//   bout[h*64+e]    = sum_d b[h*64+d]    * rel[h][d][e]   (f32)
__global__ void fuse_rel(const float* __restrict__ W, const float* __restrict__ b,
                         const float* __restrict__ rel,
                         __bf16* __restrict__ Wout, float* __restrict__ bout) {
    int k = blockIdx.x;      // 0..127 (input channel)
    int n = threadIdx.x;     // 0..127 (output channel)
    int h = n >> 6, e = n & 63;
    const float* wr = W + (size_t)k * CC + h * DD;
    const float* rr = rel + (size_t)h * DD * DD + e;   // rel[h][d][e], stride DD over d
    float acc = 0.f;
    #pragma unroll 8
    for (int d = 0; d < DD; ++d) acc += wr[d] * rr[(size_t)d * DD];
    Wout[(size_t)k * CC + n] = (__bf16)acc;
    if (k == 0) {
        const float* br = b + h * DD;
        float bacc = 0.f;
        for (int d = 0; d < DD; ++d) bacc += br[d] * rr[(size_t)d * DD];
        bout[n] = bacc;
    }
}

// ---------------------------------------------------------------------------
// WMMA GEMM: out[M,128] = epilogue( A[M,128](f32) @ W[128,128](bf16) + bias )
// Block: 256 threads (8 wave32s), tile 128 rows x 128 cols, K chunked by 32.
// Weight tile (32 KB bf16) staged global->LDS via CDNA5 async copy (ASYNCcnt);
// A tile staged via VALU (needs f32->bf16 conversion anyway).
// ---------------------------------------------------------------------------
#define SW_ROW_BYTES 272   // 128 bf16 cols * 2B + 16B pad (16B aligned)

template <bool IN_GELU, bool OUT_RELU, bool SKIP_BLEND, bool OUT_BF16>
__global__ void __launch_bounds__(256)
gemm128(const float* __restrict__ A, int M,
        const __bf16* __restrict__ W, const float* __restrict__ bias,
        float* __restrict__ outF, __bf16* __restrict__ outB,
        const float* __restrict__ x_old, const float* __restrict__ skip_scalar) {
    __shared__ __bf16 sA[128][34];                   // 128 rows x 32 k (padded)
    __shared__ __bf16 sW[128][SW_ROW_BYTES / 2];     // full 128x128 W (padded rows)

    const int tid  = threadIdx.x;
    const int wave = tid >> 5;
    const int lane = tid & 31;
    const int blockRow = blockIdx.x * 128;

    // ---- kick off async global->LDS copy of the full weight tile ----
    // 128 rows x 256 bytes = 2048 x 16B chunks; 256 threads x 8 ops.
    {
        const unsigned sWbase = (unsigned)(uintptr_t)(&sW[0][0]);
        #pragma unroll
        for (int i = 0; i < 8; ++i) {
            int chunk   = i * 256 + tid;         // 16-byte chunk id, 0..2047
            int byteoff = chunk << 4;            // global byte offset
            int krow    = byteoff >> 8;          // /256B per row
            int colb    = byteoff & 255;         // byte within row
            unsigned ldsa = sWbase + (unsigned)(krow * SW_ROW_BYTES + colb);
            const char* gp = (const char*)W + byteoff;
            asm volatile("global_load_async_to_lds_b128 %0, %1, off"
                         :: "v"(ldsa), "v"(gp) : "memory");
        }
    }

    v8f acc[8];
    #pragma unroll
    for (int t = 0; t < 8; ++t)
        #pragma unroll
        for (int j = 0; j < 8; ++j) acc[t][j] = 0.0f;

    for (int kc = 0; kc < 4; ++kc) {
        // ---- stage A chunk (f32 -> bf16) while the async W copy flies ----
        {
            int r  = tid >> 1;
            int k0 = (tid & 1) * 16;
            int gr = blockRow + r;
            const float* ap = A + (size_t)gr * CC + kc * 32 + k0;
            if (gr < M && kc < 3) __builtin_prefetch(ap + 32, 0, 0);
            #pragma unroll
            for (int i = 0; i < 16; i += 4) {
                float4 f;
                if (gr < M) f = *(const float4*)(ap + i);
                else        f = make_float4(0.f, 0.f, 0.f, 0.f);
                if (IN_GELU) {
                    f.x = gelu_exact(f.x); f.y = gelu_exact(f.y);
                    f.z = gelu_exact(f.z); f.w = gelu_exact(f.w);
                }
                sA[r][k0 + i + 0] = (__bf16)f.x;
                sA[r][k0 + i + 1] = (__bf16)f.y;
                sA[r][k0 + i + 2] = (__bf16)f.z;
                sA[r][k0 + i + 3] = (__bf16)f.w;
            }
        }
        if (kc == 0) {
            // each wave waits for its own async transfers, then barrier
            asm volatile("s_wait_asynccnt 0x0" ::: "memory");
        }
        __syncthreads();

        // ---- A fragment: 16x32 bf16 layout (lanes 0-15: K 0..7 / 16..23;
        //                                     lanes 16-31: K 8..15 / 24..31) ----
        v16bf afrag;
        {
            int m   = wave * 16 + (lane & 15);
            int kb0 = (lane < 16) ? 0 : 8;
            int kb1 = (lane < 16) ? 16 : 24;
            #pragma unroll
            for (int i = 0; i < 4; ++i) {
                afrag[2 * i]     = sA[m][kb0 + 2 * i];
                afrag[2 * i + 1] = sA[m][kb0 + 2 * i + 1];
                afrag[8 + 2 * i]     = sA[m][kb1 + 2 * i];
                afrag[8 + 2 * i + 1] = sA[m][kb1 + 2 * i + 1];
            }
        }
        // ---- 8 WMMAs across the 128-wide N dimension ----
        #pragma unroll
        for (int t = 0; t < 8; ++t) {
            v16bf bfrag;                      // B 32x16: lane%16 == N col, K in same order as A
            int n0 = t * 16 + (lane & 15);
            int kr0 = kc * 32 + ((lane < 16) ? 0 : 8);
            int kr1 = kc * 32 + ((lane < 16) ? 16 : 24);
            #pragma unroll
            for (int i = 0; i < 4; ++i) {
                bfrag[2 * i]         = sW[kr0 + 2 * i][n0];
                bfrag[2 * i + 1]     = sW[kr0 + 2 * i + 1][n0];
                bfrag[8 + 2 * i]     = sW[kr1 + 2 * i][n0];
                bfrag[8 + 2 * i + 1] = sW[kr1 + 2 * i + 1][n0];
            }
            acc[t] = __builtin_amdgcn_wmma_f32_16x16x32_bf16(
                false, afrag, false, bfrag, (short)0, acc[t], false, false);
        }
        __syncthreads();
    }

    // ---- epilogue ----
    float a_s = 0.f, b_s = 0.f;
    if (SKIP_BLEND) {
        float s = *skip_scalar;
        a_s = 1.0f / (1.0f + expf(-s));
        b_s = 1.0f - a_s;
    }
    #pragma unroll
    for (int t = 0; t < 8; ++t) {
        int col = t * 16 + (lane & 15);
        float bcol = bias[col];
        #pragma unroll
        for (int j = 0; j < 8; ++j) {
            int row = blockRow + wave * 16 + j + ((lane < 16) ? 0 : 8);
            if (row < M) {
                float v = acc[t][j] + bcol;
                if (OUT_RELU)   v = fmaxf(v, 0.f);
                if (SKIP_BLEND) v = a_s * v + b_s * x_old[(size_t)row * CC + col];
                if (OUT_BF16) outB[(size_t)row * CC + col] = (__bf16)v;
                else          outF[(size_t)row * CC + col] = v;
            }
        }
    }
}

// ---------------------------------------------------------------------------
// Edge phase: logits + running max (monotone uint key)
// one thread per (edge, head); 64-dot in bf16 -> f32
// ---------------------------------------------------------------------------
__global__ void edge_logit_max(const __bf16* __restrict__ q, const __bf16* __restrict__ krel,
                               const int* __restrict__ sidx, const int* __restrict__ didx,
                               int E, const float* __restrict__ prel,
                               float* __restrict__ exbuf, unsigned* __restrict__ mxk) {
    long long i = (long long)blockIdx.x * blockDim.x + threadIdx.x;
    if (i >= (long long)E * HH) return;
    int e = (int)(i >> 1), h = (int)(i & 1);
    int s = sidx[e], d = didx[e];
    const uint4* qv = (const uint4*)(q + (size_t)d * CC + h * DD);
    const uint4* kv = (const uint4*)(krel + (size_t)s * CC + h * DD);
    float acc = 0.f;
    #pragma unroll
    for (int j = 0; j < 8; ++j) {
        union { uint4 u; __bf16 b[8]; } qa, ka;
        qa.u = qv[j]; ka.u = kv[j];
        #pragma unroll
        for (int c = 0; c < 8; ++c) acc += (float)qa.b[c] * (float)ka.b[c];
    }
    float lg = acc * 0.125f * prel[h];     // scale = 1/sqrt(64)
    exbuf[i] = lg;
    atomicMax(mxk + (size_t)d * HH + h, fkey(lg));
}

// exp(l - max), accumulate denominator
__global__ void edge_exp(const int* __restrict__ didx, int E,
                         float* __restrict__ exbuf,
                         const unsigned* __restrict__ mxk, float* __restrict__ den) {
    long long i = (long long)blockIdx.x * blockDim.x + threadIdx.x;
    if (i >= (long long)E * HH) return;
    int e = (int)(i >> 1), h = (int)(i & 1);
    int d = didx[e];
    float m = funkey(mxk[(size_t)d * HH + h]);
    float v = expf(exbuf[i] - m);
    exbuf[i] = v;
    atomicAdd(den + (size_t)d * HH + h, v);
}

// weighted scatter-add of messages: 32 lanes per edge, 4 channels per lane
__global__ void __launch_bounds__(256)
edge_agg(const int* __restrict__ sidx, const int* __restrict__ didx, int E,
         const float* __restrict__ exbuf, const float* __restrict__ den,
         const __bf16* __restrict__ vrel, float* __restrict__ agg) {
    int tid = threadIdx.x;
    int lane = tid & 31;
    long long e = (long long)blockIdx.x * 8 + (tid >> 5);
    if (e >= E) return;
    int s = sidx[e], d = didx[e];
    int ch = lane * 4;
    int h = ch >> 6;
    float alpha = exbuf[e * HH + h] / den[(size_t)d * HH + h];
    union { uint2 u; __bf16 b[4]; } va;
    va.u = *(const uint2*)(vrel + (size_t)s * CC + ch);
    float* ap = agg + (size_t)d * CC + ch;
    #pragma unroll
    for (int i = 0; i < 4; ++i) atomicAdd(ap + i, alpha * (float)va.b[i]);
}

// ---------------------------------------------------------------------------
extern "C" void kernel_launch(void* const* d_in, const int* in_sizes, int n_in,
                              void* d_out, int out_size, void* d_ws, size_t ws_size,
                              hipStream_t stream) {
    const float* x_model   = (const float*)d_in[0];
    const float* x_dataset = (const float*)d_in[1];
    const int*   src_md    = (const int*)d_in[2];
    const int*   dst_md    = (const int*)d_in[3];
    const int*   src_dm    = (const int*)d_in[4];
    const int*   dst_dm    = (const int*)d_in[5];
    const float* Wl = (const float*)d_in[6];
    const float* bl = (const float*)d_in[7];
    const float* Wk = (const float*)d_in[8];
    const float* bk = (const float*)d_in[9];
    const float* Wq = (const float*)d_in[10];
    const float* bq = (const float*)d_in[11];
    const float* Wv = (const float*)d_in[12];
    const float* bv = (const float*)d_in[13];
    const float* Wa = (const float*)d_in[14];
    const float* ba = (const float*)d_in[15];
    const float* skip  = (const float*)d_in[16];
    const float* a_rel = (const float*)d_in[17];
    const float* m_rel = (const float*)d_in[18];
    const float* p_rel = (const float*)d_in[19];

    // workspace carve-out (256B aligned)
    char* base = (char*)d_ws;
    size_t off = 0;
    auto take = [&](size_t bytes) -> void* {
        void* p = base + off;
        off += (bytes + 255) & ~(size_t)255;
        return p;
    };
    float*    xs    = (float*)take((size_t)NTOT * CC * 4);
    float*    agg   = (float*)take((size_t)NTOT * CC * 4);
    __bf16*   qb    = (__bf16*)take((size_t)NTOT * CC * 2);
    __bf16*   krel  = (__bf16*)take((size_t)NTOT * CC * 2);
    __bf16*   vrel  = (__bf16*)take((size_t)NTOT * CC * 2);
    float*    exbuf = (float*)take((size_t)RR * EDGES * HH * 4);
    unsigned* mxk   = (unsigned*)take((size_t)NTOT * HH * 4);
    float*    den   = (float*)take((size_t)NTOT * HH * 4);
    __bf16*   wWl   = (__bf16*)take((size_t)TT * CC * CC * 2);
    __bf16*   wWq   = (__bf16*)take((size_t)LL * TT * CC * CC * 2);
    __bf16*   wWa   = (__bf16*)take((size_t)LL * TT * CC * CC * 2);
    __bf16*   wWk   = (__bf16*)take((size_t)LL * RR * CC * CC * 2);
    __bf16*   wWv   = (__bf16*)take((size_t)LL * RR * CC * CC * 2);
    float*    fbk   = (float*)take((size_t)LL * RR * CC * 4);
    float*    fbv   = (float*)take((size_t)LL * RR * CC * 4);

    // ---- weight prep ----
    const int cvtG = (CC * CC + 255) / 256;
    for (int t = 0; t < TT; ++t)
        cvt_bf16<<<cvtG, 256, 0, stream>>>(Wl + (size_t)t * CC * CC, wWl + (size_t)t * CC * CC, CC * CC);
    for (int l = 0; l < LL; ++l)
        for (int t = 0; t < TT; ++t) {
            size_t o = (size_t)(l * TT + t) * CC * CC;
            cvt_bf16<<<cvtG, 256, 0, stream>>>(Wq + o, wWq + o, CC * CC);
            cvt_bf16<<<cvtG, 256, 0, stream>>>(Wa + o, wWa + o, CC * CC);
        }
    for (int l = 0; l < LL; ++l)
        for (int r = 0; r < RR; ++r) {
            int st = r;   // r=0: src type 0 (model); r=1: src type 1 (dataset)
            size_t wo = (size_t)(l * TT + st) * CC * CC;
            size_t bo = (size_t)(l * TT + st) * CC;
            size_t ro = (size_t)(l * RR + r) * HH * DD * DD;
            size_t fo = (size_t)(l * RR + r) * CC * CC;
            size_t fb = (size_t)(l * RR + r) * CC;
            fuse_rel<<<CC, CC, 0, stream>>>(Wk + wo, bk + bo, a_rel + ro, wWk + fo, fbk + fb);
            fuse_rel<<<CC, CC, 0, stream>>>(Wv + wo, bv + bo, m_rel + ro, wWv + fo, fbv + fb);
        }

    const size_t offT[2] = {0, (size_t)NMODEL};
    const int    Ms[2]   = {NMODEL, NDATASET};
    const int    grids[2] = {(NMODEL + 127) / 128, (NDATASET + 127) / 128};

    // ---- input projection + ReLU ----
    gemm128<false, true, false, false><<<grids[0], 256, 0, stream>>>(
        x_model, NMODEL, wWl, bl, xs, nullptr, nullptr, nullptr);
    gemm128<false, true, false, false><<<grids[1], 256, 0, stream>>>(
        x_dataset, NDATASET, wWl + (size_t)CC * CC, bl + CC,
        xs + offT[1] * CC, nullptr, nullptr, nullptr);

    const int* sI[2] = {src_md, src_dm};
    const int* dI[2] = {dst_md, dst_dm};
    const int  dtA[2] = {1, 0};
    const int  gE = (int)(((long long)EDGES * HH + 255) / 256);
    const int  gAgg = (EDGES + 7) / 8;

    for (int l = 0; l < LL; ++l) {
        hipMemsetAsync(agg, 0, (size_t)NTOT * CC * 4, stream);
        hipMemsetAsync(mxk, 0, (size_t)NTOT * HH * 4, stream);   // key 0 == -inf
        hipMemsetAsync(den, 0, (size_t)NTOT * HH * 4, stream);

        // q projections (bf16 out)
        for (int t = 0; t < TT; ++t) {
            size_t o = (size_t)(l * TT + t);
            gemm128<false, false, false, true><<<grids[t], 256, 0, stream>>>(
                xs + offT[t] * CC, Ms[t], wWq + o * CC * CC, bq + o * CC,
                nullptr, qb + offT[t] * CC, nullptr, nullptr);
        }
        // fused k_rel / v_rel projections (bf16 out)
        for (int r = 0; r < RR; ++r) {
            int st = r;
            size_t o = (size_t)(l * RR + r);
            gemm128<false, false, false, true><<<grids[st], 256, 0, stream>>>(
                xs + offT[st] * CC, Ms[st], wWk + o * CC * CC, fbk + o * CC,
                nullptr, krel + offT[st] * CC, nullptr, nullptr);
            gemm128<false, false, false, true><<<grids[st], 256, 0, stream>>>(
                xs + offT[st] * CC, Ms[st], wWv + o * CC * CC, fbv + o * CC,
                nullptr, vrel + offT[st] * CC, nullptr, nullptr);
        }
        // edge phase
        for (int r = 0; r < RR; ++r) {
            int st = r, dt = dtA[r];
            edge_logit_max<<<gE, 256, 0, stream>>>(
                qb + offT[dt] * CC, krel + offT[st] * CC, sI[r], dI[r], EDGES,
                p_rel + (size_t)(l * RR + r) * HH,
                exbuf + (size_t)r * EDGES * HH, mxk + offT[dt] * HH);
        }
        for (int r = 0; r < RR; ++r) {
            int dt = dtA[r];
            edge_exp<<<gE, 256, 0, stream>>>(
                dI[r], EDGES, exbuf + (size_t)r * EDGES * HH,
                mxk + offT[dt] * HH, den + offT[dt] * HH);
        }
        for (int r = 0; r < RR; ++r) {
            int st = r, dt = dtA[r];
            edge_agg<<<gAgg, 256, 0, stream>>>(
                sI[r], dI[r], EDGES, exbuf + (size_t)r * EDGES * HH,
                den + offT[dt] * HH, vrel + offT[st] * CC, agg + offT[dt] * CC);
        }
        // output projection: gelu(agg) @ Wa + ba, sigmoid-skip blend
        for (int t = 0; t < TT; ++t) {
            size_t o = (size_t)(l * TT + t);
            float* outp = (l == LL - 1) ? ((float*)d_out + offT[t] * CC)
                                        : (xs + offT[t] * CC);
            gemm128<true, false, true, false><<<grids[t], 256, 0, stream>>>(
                agg + offT[t] * CC, Ms[t], wWa + o * CC * CC, ba + o * CC,
                outp, nullptr, xs + offT[t] * CC, skip + (l * TT + t));
        }
    }
    (void)in_sizes; (void)n_in; (void)out_size; (void)ws_size;
}
